// GroupedQueryAttention_36799279792765
// MI455X (gfx1250) — matrix operand order repaired and checked by reading, not compiled, not run on hardware
//
#include <hip/hip_runtime.h>
#include <hip/hip_bf16.h>

// GQA forward for MI455X (gfx1250): bf16 WMMA (v_wmma_f32_16x16x32_bf16),
// f32 accumulate, flash-style online-softmax attention, wave32 layouts per
// CDNA5 ISA. GEMM tiles use GLOBAL_LOAD_ASYNC_TO_LDS (ASYNCcnt)
// when the toolchain exposes the builtins, else aligned LDS stores.

typedef __attribute__((ext_vector_type(16))) __bf16 v16bf;
typedef __attribute__((ext_vector_type(8)))  __bf16 v8bf;
typedef __attribute__((ext_vector_type(4)))  __bf16 v4bf;
typedef __attribute__((ext_vector_type(8)))  float  v8f;

#define T_SEQ   2048
#define D_MODEL 2048
#define NH      32
#define NKVH    8
#define GQA_G   4
#define HD      64

#if defined(__AMDGCN__) && __has_builtin(__builtin_amdgcn_global_load_async_to_lds_b128) && __has_builtin(__builtin_amdgcn_s_wait_asynccnt)
#define USE_ASYNC_LDS 1
typedef int v4i_ __attribute__((ext_vector_type(4)));
typedef __attribute__((address_space(1))) v4i_* as1_v4i;   // global source
typedef __attribute__((address_space(3))) v4i_* as3_v4i;   // LDS destination
#else
#define USE_ASYNC_LDS 0
#endif

static __device__ __forceinline__ v8f wmma_bf16(const v16bf& a, const v16bf& b, const v8f& c) {
    return __builtin_amdgcn_wmma_f32_16x16x32_bf16(false, a, false, b, (short)0, c, false, false);
}

// ---------------------------------------------------------------------------
// fp32 -> bf16 (vectorized, 4 elems/thread)
// ---------------------------------------------------------------------------
__global__ __launch_bounds__(256)
void to_bf16_kernel(const float* __restrict__ in, __bf16* __restrict__ out, int n4)
{
    const int i = blockIdx.x * 256 + threadIdx.x;
    if (i < n4) {
        const float4 v = ((const float4*)in)[i];
        v4bf o;
        o[0] = (__bf16)v.x; o[1] = (__bf16)v.y; o[2] = (__bf16)v.z; o[3] = (__bf16)v.w;
        ((v4bf*)out)[i] = o;
    }
}

// ---------------------------------------------------------------------------
// fp32 [K][N] -> bf16 transposed [N][K]  (32x32 LDS tile transpose)
// ---------------------------------------------------------------------------
__global__ __launch_bounds__(256)
void transpose_to_bf16_kernel(const float* __restrict__ W, __bf16* __restrict__ Wt,
                              int K, int N)
{
    __shared__ __bf16 tile[32][33];
    const int nb = blockIdx.x * 32;
    const int kb = blockIdx.y * 32;
    const int tx = threadIdx.x & 31;
    const int ty = threadIdx.x >> 5;          // 0..7
    #pragma unroll
    for (int i = 0; i < 32; i += 8)
        tile[ty + i][tx] = (__bf16)W[(size_t)(kb + ty + i) * N + nb + tx];
    __syncthreads();
    #pragma unroll
    for (int i = 0; i < 32; i += 8)
        Wt[(size_t)(nb + ty + i) * K + kb + tx] = tile[tx][ty + i];
}

// ---------------------------------------------------------------------------
// Tiled GEMM:  Out[M,N] = A[M,K] * Bt[N,K]^T   (both operands bf16 in HBM)
//   ROPE      : interleaved RoPE per 64-wide head in the epilogue
//   TRANS_OUT : store bf16 output transposed ([N][M]) -- used for V
//   OUT_F32   : store fp32 (final output) instead of bf16
// Block: 256 threads (8 waves). Tile 64x64, K-step 32.
// Both LDS tiles are row-per-(m|n), contiguous in K -> b128 async copies.
// ---------------------------------------------------------------------------
template <bool ROPE, bool TRANS_OUT, bool OUT_F32>
__global__ __launch_bounds__(256)
void gemm_wmma(const __bf16* __restrict__ A,   // [M][K]
               const __bf16* __restrict__ Bt,  // [N][K]
               const float* __restrict__ freqs, void* __restrict__ Out,
               int M, int N, int K, float out_scale)
{
    __shared__ __bf16 As[64][40];   // [m][k] (row stride 80B, 16B-aligned segs)
    __shared__ __bf16 Bs[64][40];   // [n][k]

    const int tid    = threadIdx.x;
    const int lane   = tid & 31;
    const int half16 = lane >> 4;
    const int l15    = lane & 15;
    const int wave   = tid >> 5;

    const int m0 = blockIdx.y * 64;
    const int n0 = blockIdx.x * 64;

    const int mi     = wave & 3;          // m-subtile index 0..3
    const int njbase = (wave >> 2) * 2;   // first of two n-subtiles

    v8f acc0 = {}; v8f acc1 = {};

    const int lrow = tid >> 2;            // 0..63 (tile row for this thread)
    const int lseg = (tid & 3) * 8;       // K-segment: 0,8,16,24 (8 halfs = 16B)

    const __bf16* gA = A  + (size_t)(m0 + lrow) * K + lseg;
    const __bf16* gB = Bt + (size_t)(n0 + lrow) * K + lseg;

    for (int k0 = 0; k0 < K; k0 += 32) {
#if USE_ASYNC_LDS
        __builtin_amdgcn_global_load_async_to_lds_b128(
            (as1_v4i)(gA + k0), (as3_v4i)&As[lrow][lseg], 0, 0);
        __builtin_amdgcn_global_load_async_to_lds_b128(
            (as1_v4i)(gB + k0), (as3_v4i)&Bs[lrow][lseg], 0, 0);
        __builtin_amdgcn_s_wait_asynccnt(0);
#else
        *(v8bf*)&As[lrow][lseg] = *(const v8bf*)(gA + k0);
        *(v8bf*)&Bs[lrow][lseg] = *(const v8bf*)(gB + k0);
        if (k0 + 32 < K) {   // prefetch next tiles (global_prefetch_b8)
            __builtin_prefetch(gA + k0 + 32, 0, 3);
            __builtin_prefetch(gB + k0 + 32, 0, 3);
        }
#endif
        __syncthreads();

        // ---- A fragment (16x32, M rows per lane; contiguous K = 16*half + e) ----
        v16bf a;
        {
            const int m  = mi * 16 + l15;
            const int kb = half16 * 16;
            #pragma unroll
            for (int e = 0; e < 8; ++e)  a[e] = As[m][kb + e];       // K = kb+e
            #pragma unroll
            for (int e = 8; e < 16; ++e) a[e] = As[m][kb + e];       // K = kb+e
        }
        // ---- B fragments (32x16, K rows striped across halves) ----
        {
            const int kb = half16 * 16;
            const int na = (njbase + 0) * 16 + l15;
            const int nb = (njbase + 1) * 16 + l15;
            v16bf b0, b1;
            #pragma unroll
            for (int e = 0; e < 16; ++e) b0[e] = Bs[na][kb + e];
            #pragma unroll
            for (int e = 0; e < 16; ++e) b1[e] = Bs[nb][kb + e];
            acc0 = wmma_bf16(a, b0, acc0);
            acc1 = wmma_bf16(a, b1, acc1);
        }
        __syncthreads();
    }

    // ---- epilogue: optional RoPE, scale, store ----
    #pragma unroll
    for (int tt = 0; tt < 2; ++tt) {
        v8f acc = tt ? acc1 : acc0;
        const int nj   = njbase + tt;
        const int ncol = n0 + nj * 16 + l15;      // D-layout: N = lane&15
        #pragma unroll
        for (int r = 0; r < 8; ++r) {
            const int row = m0 + mi * 16 + half16 * 8 + r;  // D-layout: M = r + 8*half
            float v = acc[r];
            if (ROPE) {
                // even/odd columns sit in adjacent lanes in the C/D layout
                float other = __shfl_xor(v, 1, 32);
                const int i = (ncol & (HD - 1)) >> 1;
                float ang = freqs[row * (HD / 2) + i];
                float sn, cs;
                __sincosf(ang, &sn, &cs);
                v = ((ncol & 1) == 0) ? (v * cs - other * sn)
                                      : (other * sn + v * cs);
            }
            v *= out_scale;
            if (OUT_F32) {
                ((float*)Out)[(size_t)row * N + ncol] = v;
            } else if (TRANS_OUT) {
                ((__bf16*)Out)[(size_t)ncol * M + row] = (__bf16)v;
            } else {
                ((__bf16*)Out)[(size_t)row * N + ncol] = (__bf16)v;
            }
        }
    }
}

// ---------------------------------------------------------------------------
// Flash-style causal GQA attention. One wave (32 thr) per (head, 16-row tile).
//   Q   : [T][NH*HD]   bf16, already scaled by 1/sqrt(HD)
//   Kb  : [T][NKVH*HD] bf16 (RoPE'd)
//   Vt  : [NKVH*HD][T] bf16 (transposed at projection time)
//   Ctx : [T][NH*HD]   bf16
// ---------------------------------------------------------------------------
__global__ __launch_bounds__(32)
void attn_kernel(const __bf16* __restrict__ Q, const __bf16* __restrict__ Kb,
                 const __bf16* __restrict__ Vt, __bf16* __restrict__ Ctx)
{
    __shared__ __bf16 Ps[16][40];    // P tile staging (C-layout -> A-layout)

    const int lane   = threadIdx.x & 31;
    const int half16 = lane >> 4;
    const int l15    = lane & 15;
    const int h      = blockIdx.x;        // query head
    const int tb     = blockIdx.y * 16;   // q-tile base row
    const int kvh    = h / GQA_G;         // jnp.repeat(k, G, axis=2) -> h//G

    // --- preload Q fragments (A-layout), HD=64 -> two K=32 chunks ---
    v16bf aq0, aq1;
    {
        const __bf16* qrow = Q + (size_t)(tb + l15) * (NH * HD) + h * HD;
        const int kb = half16 * 16;
        #pragma unroll
        for (int e = 0; e < 8; ++e)  { aq0[e] = qrow[kb + e];      aq1[e] = qrow[32 + kb + e]; }
        #pragma unroll
        for (int e = 8; e < 16; ++e) { aq0[e] = qrow[kb + e];  aq1[e] = qrow[32 + kb + e]; }
    }

    v8f acc[4] = {{}, {}, {}, {}};       // 16(t) x 64(hd) output accumulator
    float m_i[8], l_i[8];
    #pragma unroll
    for (int r = 0; r < 8; ++r) { m_i[r] = -__builtin_inff(); l_i[r] = 0.f; }

    const int s_end = tb + 15;           // causal limit (inclusive)
    for (int sb = 0; sb <= s_end; sb += 32) {
        // ---- scores: two 16x16 subtiles, each K=64 over head dim ----
        v8f s0 = {}, s1 = {};
        {
            const int kb = half16 * 16;
            const __bf16* kr0 = Kb + (size_t)(sb + l15)      * (NKVH * HD) + kvh * HD;
            const __bf16* kr1 = Kb + (size_t)(sb + 16 + l15) * (NKVH * HD) + kvh * HD;
            v16bf b;
            #pragma unroll
            for (int e = 0; e < 16; ++e) b[e] = kr0[kb + e];
            s0 = wmma_bf16(aq0, b, s0);
            #pragma unroll
            for (int e = 0; e < 16; ++e) b[e] = kr0[32 + kb + e];
            s0 = wmma_bf16(aq1, b, s0);
            #pragma unroll
            for (int e = 0; e < 16; ++e) b[e] = kr1[kb + e];
            s1 = wmma_bf16(aq0, b, s1);
            #pragma unroll
            for (int e = 0; e < 16; ++e) b[e] = kr1[32 + kb + e];
            s1 = wmma_bf16(aq1, b, s1);
        }

        // ---- causal mask + online softmax (rows live in fixed lane halves) ----
        const int sc0 = sb + l15;
        const int sc1 = sb + 16 + l15;
        float corr[8];
        #pragma unroll
        for (int r = 0; r < 8; ++r) {
            const int trow = tb + half16 * 8 + r;
            float v0 = (sc0 > trow) ? -__builtin_inff() : s0[r];
            float v1 = (sc1 > trow) ? -__builtin_inff() : s1[r];
            float mx = fmaxf(v0, v1);
            #pragma unroll
            for (int off = 1; off < 16; off <<= 1)
                mx = fmaxf(mx, __shfl_xor(mx, off, 32));
            const float mnew = fmaxf(m_i[r], mx);
            corr[r] = __expf(m_i[r] - mnew);
            m_i[r]  = mnew;
            const float p0 = __expf(v0 - mnew);   // exp(-inf)=0 handles mask
            const float p1 = __expf(v1 - mnew);
            float rs = p0 + p1;
            #pragma unroll
            for (int off = 1; off < 16; off <<= 1)
                rs += __shfl_xor(rs, off, 32);
            l_i[r] = l_i[r] * corr[r] + rs;
            // stage P (bf16) to LDS: row = M, col = s within chunk
            Ps[half16 * 8 + r][l15]      = (__bf16)p0;
            Ps[half16 * 8 + r][16 + l15] = (__bf16)p1;
            // rescale existing accumulator rows
            #pragma unroll
            for (int j = 0; j < 4; ++j) acc[j][r] *= corr[r];
        }
        __builtin_amdgcn_wave_barrier();   // order LDS stores vs loads (in-order per wave)

        // ---- P fragment (A-layout 16x32) ----
        v16bf pa;
        {
            const int kb = half16 * 16;
            #pragma unroll
            for (int e = 0; e < 8; ++e)  pa[e] = Ps[l15][kb + e];
            #pragma unroll
            for (int e = 8; e < 16; ++e) pa[e] = Ps[l15][kb + e];
        }
        // ---- acc += P * V  (V transposed: contiguous per-lane B reads) ----
        {
            const int kb = half16 * 16;
            #pragma unroll
            for (int j = 0; j < 4; ++j) {
                const __bf16* vr = Vt + (size_t)(kvh * HD + j * 16 + l15) * T_SEQ + sb + kb;
                v16bf b;
                #pragma unroll
                for (int e = 0; e < 16; ++e) b[e] = vr[e];
                acc[j] = wmma_bf16(pa, b, acc[j]);
            }
        }
        __builtin_amdgcn_wave_barrier();
    }

    // ---- normalize and store context (bf16) ----
    #pragma unroll
    for (int r = 0; r < 8; ++r) {
        const float inv = 1.0f / l_i[r];
        const int trow = tb + half16 * 8 + r;
        #pragma unroll
        for (int j = 0; j < 4; ++j)
            Ctx[(size_t)trow * (NH * HD) + h * HD + j * 16 + l15] = (__bf16)(acc[j][r] * inv);
    }
}

// ---------------------------------------------------------------------------
extern "C" void kernel_launch(void* const* d_in, const int* in_sizes, int n_in,
                              void* d_out, int out_size, void* d_ws, size_t ws_size,
                              hipStream_t stream)
{
    const float* x     = (const float*)d_in[0];   // [1][2048][2048]
    const float* freqs = (const float*)d_in[1];   // [2048][32]
    const float* wq    = (const float*)d_in[2];   // [2048][2048]
    const float* wk    = (const float*)d_in[3];   // [2048][512]
    const float* wv    = (const float*)d_in[4];   // [2048][512]
    const float* wo    = (const float*)d_in[5];   // [2048][2048]
    (void)in_sizes; (void)n_in; (void)out_size; (void)ws_size;

    // workspace layout (bf16), total 24M elems = 48MB:
    // xb 4M | wqt 4M | wkt 1M | wvt 1M | wot 4M | qb 4M | kb 1M | vt 1M | ctx 4M
    __bf16* xb  = (__bf16*)d_ws;
    __bf16* wqt = xb  + (size_t)T_SEQ * D_MODEL;
    __bf16* wkt = wqt + (size_t)D_MODEL * (NH * HD);
    __bf16* wvt = wkt + (size_t)D_MODEL * (NKVH * HD);
    __bf16* wot = wvt + (size_t)D_MODEL * (NKVH * HD);
    __bf16* qb  = wot + (size_t)(NH * HD) * D_MODEL;
    __bf16* kb  = qb  + (size_t)T_SEQ * NH * HD;
    __bf16* vt  = kb  + (size_t)T_SEQ * NKVH * HD;
    __bf16* ctx = vt  + (size_t)T_SEQ * NKVH * HD;

    const dim3 blk(256);

    // --- pre-pass: bf16 conversions (x) and bf16+transpose (weights) ---
    to_bf16_kernel<<<dim3((T_SEQ * D_MODEL / 4 + 255) / 256), blk, 0, stream>>>(
        x, xb, T_SEQ * D_MODEL / 4);
    transpose_to_bf16_kernel<<<dim3((NH * HD) / 32, D_MODEL / 32), blk, 0, stream>>>(
        wq, wqt, D_MODEL, NH * HD);
    transpose_to_bf16_kernel<<<dim3((NKVH * HD) / 32, D_MODEL / 32), blk, 0, stream>>>(
        wk, wkt, D_MODEL, NKVH * HD);
    transpose_to_bf16_kernel<<<dim3((NKVH * HD) / 32, D_MODEL / 32), blk, 0, stream>>>(
        wv, wvt, D_MODEL, NKVH * HD);
    transpose_to_bf16_kernel<<<dim3(D_MODEL / 32, (NH * HD) / 32), blk, 0, stream>>>(
        wo, wot, NH * HD, D_MODEL);

    // --- projections ---
    // Q = rope(x@wq) * 1/sqrt(HD)
    gemm_wmma<true, false, false>
        <<<dim3((NH * HD) / 64, T_SEQ / 64), blk, 0, stream>>>(
            xb, wqt, freqs, qb, T_SEQ, NH * HD, D_MODEL, 0.125f);
    // K = rope(x@wk)
    gemm_wmma<true, false, false>
        <<<dim3((NKVH * HD) / 64, T_SEQ / 64), blk, 0, stream>>>(
            xb, wkt, freqs, kb, T_SEQ, NKVH * HD, D_MODEL, 1.0f);
    // V^T = (x@wv)^T
    gemm_wmma<false, true, false>
        <<<dim3((NKVH * HD) / 64, T_SEQ / 64), blk, 0, stream>>>(
            xb, wvt, freqs, vt, T_SEQ, NKVH * HD, D_MODEL, 1.0f);

    // --- attention ---
    attn_kernel<<<dim3(NH, T_SEQ / 16), dim3(32), 0, stream>>>(qb, kb, vt, ctx);

    // --- output projection (fp32 out) ---
    gemm_wmma<false, false, true>
        <<<dim3(D_MODEL / 64, T_SEQ / 64), blk, 0, stream>>>(
            ctx, wot, freqs, d_out, T_SEQ, D_MODEL, NH * HD, 1.0f);
}
